// UpscaleRAFT_65214783422642
// MI455X (gfx1250) — matrix operand order repaired and checked by reading, not compiled, not run on hardware
//
#include <hip/hip_runtime.h>
#include <hip/hip_bf16.h>

// ---------- types for WMMA fragments ----------
typedef __attribute__((ext_vector_type(16))) __bf16 v16bf;
typedef __attribute__((ext_vector_type(8)))  float  v8f;
typedef __attribute__((ext_vector_type(4)))  unsigned int u32x4;
typedef __attribute__((ext_vector_type(8)))  int          i32x8;
typedef __attribute__((ext_vector_type(4)))  int          i32x4;

union Frag16 { v16bf v; uint4 q[2]; };   // 32 bytes

__device__ __forceinline__ unsigned short f32_to_bf16(float f) {
    unsigned int u = __float_as_uint(f);
    unsigned int r = u + 0x7FFFu + ((u >> 16) & 1u);       // round-to-nearest-even
    if ((u & 0x7F800000u) == 0x7F800000u) r = u;           // inf/nan: truncate
    return (unsigned short)(r >> 16);
}

// ---------- constants ----------
#define NB    8
#define CIN   512
#define HH    96
#define WW    96
#define HWSZ  9216             // 96*96
#define MTOT  73728            // 8*96*96
#define K1    4608             // 512*9
#define CO1   256
#define CO2   576
#define XSTR  264              // padded x-row stride (528B = 132 words, 132%64=4 -> no LDS bank conflicts)
#define PH    98               // padded H
#define PW    98               // padded W
#define PIMG  (PH * PW)        // 9604
#define FMPAD_ELTS ((size_t)NB * PIMG * CIN)   // 39,337,984 bf16

// ---------- Tensor Data Mover availability ----------
#if defined(__has_builtin)
#  if __has_builtin(__builtin_amdgcn_tensor_load_to_lds)
#    define HAVE_TDM 1
#  endif
#endif
#ifndef HAVE_TDM
#  define HAVE_TDM 0
#endif
#if __has_include(<hip/amd_detail/amd_gfx1250_TDM.h>)
#  define TDM_6ARG 1           // therock-10.0 headers -> 6-arg builtin
#else
#  define TDM_6ARG 0           // ROCm 7.2 -> 5-arg builtin
#endif

#if HAVE_TDM
// 2-D tile load: global (bf16, row-major, stride strideD0 elems) -> LDS (dense)
// Descriptor layout per CDNA5 ISA.
__device__ __forceinline__ void tdm_load_tile_2d(
    unsigned ldsAddr, const void* gptr,
    unsigned tileD0, unsigned tileD1,
    unsigned long long tensorD0, unsigned long long tensorD1,
    unsigned long long strideD0)
{
    const unsigned long long ga = (unsigned long long)gptr;
    u32x4 g0;
    g0.x = 1u;                                            // count=1 (valid user desc)
    g0.y = ldsAddr;                                       // LDS byte address
    g0.z = (unsigned)(ga & 0xFFFFFFFFull);                // global_addr[31:0]
    g0.w = (unsigned)((ga >> 32) & 0x1FFFFFFull)          // global_addr[56:32]
         | (2u << 30);                                    // type = 2 ("image")
    i32x8 g1;
    g1[0] = 0x00010000;                                   // data_size=1 (2 bytes), no mask/pad/iter
    g1[1] = (int)((tensorD0 & 0xFFFFull) << 16);          // tensor_dim0[15:0] at bits 63:48
    g1[2] = (int)(((tensorD0 >> 16) & 0xFFFFull)          // tensor_dim0[31:16]
          | ((tensorD1 & 0xFFFFull) << 16));              // tensor_dim1[15:0]
    g1[3] = (int)(((tensorD1 >> 16) & 0xFFFFull)          // tensor_dim1[31:16]
          | ((unsigned long long)tileD0 << 16));          // tile_dim0
    g1[4] = (int)(tileD1 & 0xFFFFu);                      // tile_dim1 (tile_dim2 = 0)
    g1[5] = (int)(strideD0 & 0xFFFFFFFFull);              // tensor_dim0_stride[31:0]
    g1[6] = (int)((strideD0 >> 32) & 0xFFFFull);          // stride[47:32]; dim1_stride=0
    g1[7] = 0;
    const i32x4 gz = {0, 0, 0, 0};
#if TDM_6ARG
    const i32x8 gz8 = {0, 0, 0, 0, 0, 0, 0, 0};
    __builtin_amdgcn_tensor_load_to_lds(g0, g1, gz, gz, gz8, 0);
#else
    __builtin_amdgcn_tensor_load_to_lds(g0, g1, gz, gz, 0);
#endif
}
#endif

// ============================================================
// Prep 0: zero-fill padded feature buffer (uint4 = 8 bf16)
// ============================================================
__global__ __launch_bounds__(256) void zero_fill(uint4* __restrict__ p, long long n4)
{
    long long i = (long long)blockIdx.x * 256 + threadIdx.x;
    if (i < n4) p[i] = make_uint4(0u, 0u, 0u, 0u);
}

// ============================================================
// Prep 1: feature_map NCHW f32 -> padded NHWC bf16 (tiled transpose)
// writes interior of [8][98][98][512]; grid (288, 16, 8), block (32, 8)
// ============================================================
__global__ __launch_bounds__(256) void fm_to_nhwc_bf16(
    const float* __restrict__ in, unsigned short* __restrict__ out)
{
    __shared__ float tile[32][33];
    const int n  = blockIdx.z;
    const int c0 = blockIdx.y * 32;
    const int s0 = blockIdx.x * 32;        // linear h*96+w
    const int tx = threadIdx.x, ty = threadIdx.y;

    #pragma unroll
    for (int i = 0; i < 32; i += 8)
        tile[ty + i][tx] = in[((size_t)n * CIN + (c0 + ty + i)) * HWSZ + (s0 + tx)];
    __syncthreads();
    #pragma unroll
    for (int i = 0; i < 32; i += 8) {
        const int s = s0 + ty + i;
        const int h = s / WW, w = s % WW;
        out[((size_t)n * PIMG + (size_t)(h + 1) * PW + (w + 1)) * CIN + (c0 + tx)] =
            f32_to_bf16(tile[tx][ty + i]);
    }
}

// ============================================================
// Prep 2: w1 OIHW -> bf16 [co][ (ki*3+kj)*512 + ci ]
// ============================================================
__global__ __launch_bounds__(256) void pack_w1(
    const float* __restrict__ w1, unsigned short* __restrict__ w1t)
{
    int idx = blockIdx.x * 256 + threadIdx.x;
    if (idx >= CO1 * K1) return;
    int co = idx / K1;
    int k  = idx % K1;
    int t  = k / CIN;
    int ci = k % CIN;
    w1t[idx] = f32_to_bf16(w1[(size_t)(co * CIN + ci) * 9 + t]);
}

// ============================================================
// Prep 3: w2 [576][256] f32 -> bf16 (layout already matches)
// ============================================================
__global__ __launch_bounds__(256) void pack_w2(
    const float* __restrict__ w2, unsigned short* __restrict__ w2t)
{
    int idx = blockIdx.x * 256 + threadIdx.x;
    if (idx >= CO2 * CO1) return;
    w2t[idx] = f32_to_bf16(w2[idx]);
}

// ============================================================
// GEMM1: implicit conv3x3 + bias + ReLU on zero-padded NHWC bf16.
// Block = 256 threads = 8 waves; block tile 64(M) x 256(N);
// wave tile 32x64 = 2x4 WMMA 16x16 accumulators.
// Branch-free inner loop: taps are compile-time immediate offsets.
// ============================================================
__global__ __launch_bounds__(256) void conv3x3_wmma(
    const unsigned short* __restrict__ fm,   // [8][98][98][512] bf16, zero border
    const unsigned short* __restrict__ w1t,  // [256][4608] bf16
    const float* __restrict__ b1,
    unsigned short* __restrict__ xo)         // [73728][XSTR] bf16
{
    const int lane = threadIdx.x & 31;
    const int wv   = threadIdx.x >> 5;
    const int mW   = wv & 1;                 // 2 M-groups
    const int nW   = wv >> 1;                // 4 N-groups
    const int m0   = blockIdx.x * 64 + mW * 32;
    const int co0  = nW * 64;
    const int ln15 = lane & 15;
    const int kbA  = (lane >= 16) ? 8 : 0;
    const int kbB  = (lane >= 16) ? 8 : 0;

    // per-lane base pointers for tap (ki=0,kj=0) in padded coords
    const unsigned short* aP[2];
    #pragma unroll
    for (int mt = 0; mt < 2; ++mt) {
        const int m = m0 + mt * 16 + ln15;
        const int n = m / HWSZ;
        const int r = m % HWSZ;
        const int h = r / WW, w = r % WW;
        aP[mt] = fm + ((size_t)n * PIMG + (size_t)h * PW + w) * CIN + kbA;
    }

    v8f acc[2][4];
    #pragma unroll
    for (int mt = 0; mt < 2; ++mt)
        #pragma unroll
        for (int nt = 0; nt < 4; ++nt)
            #pragma unroll
            for (int i = 0; i < 8; ++i) acc[mt][nt][i] = 0.f;

    const unsigned short* bRow[4];
    #pragma unroll
    for (int nt = 0; nt < 4; ++nt)
        bRow[nt] = w1t + (size_t)(co0 + nt * 16 + ln15) * K1 + kbB;

    for (int cb = 0; cb < 16; ++cb) {
        const int ci0 = cb * 32;
        #pragma unroll
        for (int t = 0; t < 9; ++t) {
            const int aOff = ((t / 3) * PW + (t % 3)) * CIN;   // constexpr after unroll
            const int kOff = t * CIN + ci0;
            Frag16 a[2];
            #pragma unroll
            for (int mt = 0; mt < 2; ++mt) {
                a[mt].q[0] = *(const uint4*)(aP[mt] + aOff + ci0);
                a[mt].q[1] = *(const uint4*)(aP[mt] + aOff + ci0 + 16);
            }
            Frag16 b[4];
            #pragma unroll
            for (int nt = 0; nt < 4; ++nt) {
                b[nt].q[0] = *(const uint4*)(bRow[nt] + kOff);
                b[nt].q[1] = *(const uint4*)(bRow[nt] + kOff + 16);
            }
            #pragma unroll
            for (int nt = 0; nt < 4; ++nt) {
                #pragma unroll
                for (int mt = 0; mt < 2; ++mt)
                    acc[mt][nt] = __builtin_amdgcn_wmma_f32_16x16x32_bf16(
                        false, a[mt].v, false, b[nt].v, (short)0, acc[mt][nt],
                        false, false);
            }
        }
    }

    // epilogue: bias + ReLU -> bf16 [m][XSTR]
    const int mHalf = (lane >= 16) ? 8 : 0;
    #pragma unroll
    for (int nt = 0; nt < 4; ++nt) {
        const int   co   = co0 + nt * 16 + ln15;
        const float bias = b1[co];
        #pragma unroll
        for (int mt = 0; mt < 2; ++mt) {
            const int mBase = m0 + mt * 16 + mHalf;
            #pragma unroll
            for (int v = 0; v < 8; ++v) {
                float val = fmaxf(acc[mt][nt][v] + bias, 0.f);
                xo[(size_t)(mBase + v) * XSTR + co] = f32_to_bf16(val);
            }
        }
    }
}

// ============================================================
// GEMM2 fused: 1x1 conv (K=256, N=576) + bias + 0.25 scale
// + softmax over 9 taps + flow unfold + upsample scatter.
// Block = 288 threads = 9 waves; tile 16(M) x 576(N).
// A tile staged once per block into LDS by the Tensor Data Mover
// (s_wait_tensorcnt + barrier), then read as ds_load_b128 fragments.
// ============================================================
__global__ __launch_bounds__(288) void mask_upsample_wmma(
    const unsigned short* __restrict__ xin,  // [73728][XSTR] bf16
    const unsigned short* __restrict__ w2t,  // [576][256] bf16
    const float* __restrict__ b2,
    const float* __restrict__ flow,          // [8][2][96][96] f32
    float* __restrict__ out)                 // [8][2][768][768] f32
{
    __shared__ unsigned short sA[16 * XSTR]; // 8448 B, conflict-free row stride
    __shared__ float sMask[16 * 592];
    __shared__ float sFlow[16][2][9];

    const int tid  = threadIdx.x;
    const int lane = tid & 31;
    const int wv   = tid >> 5;               // 0..8
    const int ln15 = lane & 15;
    const int m0   = blockIdx.x * 16;
    const int n0w  = wv * 64;
    const int kbA  = (lane >= 16) ? 8 : 0;
    const int kbB  = (lane >= 16) ? 8 : 0;

    // stage A tile [16][XSTR] into LDS
    if (wv == 0) {
#if HAVE_TDM
        tdm_load_tile_2d((unsigned)(unsigned long long)(uintptr_t)sA,
                         xin + (size_t)m0 * XSTR,
                         XSTR, 16,          // tile 264 x 16
                         XSTR, MTOT,        // tensor dims
                         XSTR);             // row stride (elements)
        __builtin_amdgcn_s_wait_tensorcnt(0);
#else
        const uint4* src = (const uint4*)(xin + (size_t)m0 * XSTR);
        uint4* dst = (uint4*)sA;
        for (int i = lane; i < 16 * XSTR / 8; i += 32) dst[i] = src[i];
#endif
    }
    __syncthreads();

    v8f acc[4];
    #pragma unroll
    for (int nt = 0; nt < 4; ++nt)
        #pragma unroll
        for (int i = 0; i < 8; ++i) acc[nt][i] = 0.f;

    const unsigned short* aRow = sA + (size_t)ln15 * XSTR + kbA;
    const unsigned short* bRow[4];
    #pragma unroll
    for (int nt = 0; nt < 4; ++nt)
        bRow[nt] = w2t + (size_t)(n0w + nt * 16 + ln15) * CO1 + kbB;

    #pragma unroll
    for (int k0 = 0; k0 < 256; k0 += 32) {
        Frag16 a;
        a.q[0] = *(const uint4*)(aRow + k0);       // ds_load_b128
        a.q[1] = *(const uint4*)(aRow + k0 + 16);
        Frag16 b[4];
        #pragma unroll
        for (int nt = 0; nt < 4; ++nt) {
            b[nt].q[0] = *(const uint4*)(bRow[nt] + k0);
            b[nt].q[1] = *(const uint4*)(bRow[nt] + k0 + 16);
        }
        #pragma unroll
        for (int nt = 0; nt < 4; ++nt)
            acc[nt] = __builtin_amdgcn_wmma_f32_16x16x32_bf16(
                false, a.v, false, b[nt].v, (short)0, acc[nt], false, false);
    }

    // mask tile -> LDS (bias + 0.25 scale fused)
    const int mHalf = (lane >= 16) ? 8 : 0;
    #pragma unroll
    for (int nt = 0; nt < 4; ++nt) {
        const int   cc   = n0w + nt * 16 + ln15;
        const float bias = b2[cc];
        #pragma unroll
        for (int v = 0; v < 8; ++v)
            sMask[(mHalf + v) * 592 + cc] = 0.25f * (acc[nt][v] + bias);
    }

    // preload 8*flow neighborhood taps: 16 m * 2 ch * 9 taps = 288 = blockDim
    {
        const int ml  = tid / 18;
        const int rem = tid % 18;
        const int c   = rem / 9;
        const int t   = rem % 9;
        const int m = m0 + ml;
        const int n = m / HWSZ;
        const int r = m % HWSZ;
        const int h = r / WW, w = r % WW;
        const int hh = h + t / 3 - 1;
        const int ww = w + t % 3 - 1;
        float v = 0.f;
        if (hh >= 0 && hh < HH && ww >= 0 && ww < WW)
            v = 8.f * flow[(((size_t)n * 2 + c) * HH + hh) * WW + ww];
        sFlow[ml][c][t] = v;
    }
    __syncthreads();

    // softmax over taps + weighted sum + scatter: 16 m * 64 (pi,pj) tasks
    for (int task = tid; task < 1024; task += 288) {
        const int ml = task >> 6;
        const int r  = task & 63;
        float v[9];
        float mx = -3.4e38f;
        #pragma unroll
        for (int t = 0; t < 9; ++t) {
            v[t] = sMask[ml * 592 + t * 64 + r];
            mx = fmaxf(mx, v[t]);
        }
        float s = 0.f;
        #pragma unroll
        for (int t = 0; t < 9; ++t) { v[t] = __expf(v[t] - mx); s += v[t]; }
        const float inv = 1.f / s;
        float u0 = 0.f, u1 = 0.f;
        #pragma unroll
        for (int t = 0; t < 9; ++t) {
            const float p = v[t] * inv;
            u0 += p * sFlow[ml][0][t];
            u1 += p * sFlow[ml][1][t];
        }
        const int m = m0 + ml;
        const int n = m / HWSZ;
        const int rr = m % HWSZ;
        const int h = rr / WW, w = rr % WW;
        const int pi = r >> 3, pj = r & 7;
        const size_t o0 = (((size_t)n * 2) * 768 + (h * 8 + pi)) * 768 + (w * 8 + pj);
        out[o0] = u0;
        out[o0 + (size_t)768 * 768] = u1;
    }
}

// ============================================================
extern "C" void kernel_launch(void* const* d_in, const int* in_sizes, int n_in,
                              void* d_out, int out_size, void* d_ws, size_t ws_size,
                              hipStream_t stream) {
    (void)in_sizes; (void)n_in; (void)out_size; (void)ws_size;
    const float* fm   = (const float*)d_in[0];
    const float* flow = (const float*)d_in[1];
    const float* w1   = (const float*)d_in[2];
    const float* b1   = (const float*)d_in[3];
    const float* w2   = (const float*)d_in[4];
    const float* b2   = (const float*)d_in[5];
    float* out = (float*)d_out;

    unsigned short* fmb  = (unsigned short*)d_ws;            // 39,337,984 el (padded)
    unsigned short* w1t  = fmb + FMPAD_ELTS;                 // 1,179,648 el
    unsigned short* w2t  = w1t + (size_t)CO1 * K1;           // 147,456 el
    unsigned short* xbuf = w2t + (size_t)CO2 * CO1;          // 19,464,192 el (stride 264)
    // total workspace: ~120.3 MB

    const long long n4 = (long long)(FMPAD_ELTS / 8);        // uint4 count
    zero_fill<<<(unsigned)((n4 + 255) / 256), 256, 0, stream>>>((uint4*)fmb, n4);
    fm_to_nhwc_bf16<<<dim3(HWSZ / 32, CIN / 32, NB), dim3(32, 8), 0, stream>>>(fm, fmb);
    pack_w1<<<(CO1 * K1 + 255) / 256, 256, 0, stream>>>(w1, w1t);
    pack_w2<<<(CO2 * CO1 + 255) / 256, 256, 0, stream>>>(w2, w2t);
    conv3x3_wmma<<<MTOT / 64, 256, 0, stream>>>(fmb, w1t, b1, xbuf);
    mask_upsample_wmma<<<MTOT / 16, 288, 0, stream>>>(xbuf, w2t, b2, flow, out);
}
